// MultiHeadAttention_90529320665712
// MI455X (gfx1250) — compile-verified, matrix operand order checked by reading, not run on hardware
//
#include <hip/hip_runtime.h>
#include <hip/hip_bf16.h>

#define DEV __device__ __forceinline__

typedef __attribute__((ext_vector_type(16))) __bf16 v16bf;
typedef __attribute__((ext_vector_type(8)))  float  v8f;
typedef unsigned int v4u __attribute__((ext_vector_type(4)));
typedef int v8i __attribute__((ext_vector_type(8)));
typedef int v4i __attribute__((ext_vector_type(4)));

constexpr int B = 4, T = 2048, C = 768, H = 12, HS = 64;
constexpr int BT = B * T;

#if defined(__has_builtin)
#if __has_builtin(__builtin_amdgcn_tensor_load_to_lds) && \
    __has_builtin(__builtin_amdgcn_s_wait_tensorcnt)
#define HAVE_TDM 1
#endif
#endif
#ifndef HAVE_TDM
#define HAVE_TDM 0
#endif

DEV v8f wmma_bf16(v16bf a, v16bf b, v8f c) {
    return __builtin_amdgcn_wmma_f32_16x16x32_bf16(false, a, false, b,
                                                   (short)0, c, false, false);
}

// 16-bit A 16x32 / B 32x16 fragment: K runs [8*half,+8) (slots 0..3) and
// [16+8*half,+8) (slots 4..7) are contiguous -> two b128 LDS loads.
DEV v16bf load_frag(const __bf16* row, int half) {
    union { v16bf v; uint4 q[2]; } u;
    u.q[0] = *(const uint4*)(row + 8 * half);
    u.q[1] = *(const uint4*)(row + 16 + 8 * half);
    return u.v;
}

DEV void cvt_store8(__bf16* dst, const float* src) {
    float4 f0 = *(const float4*)src;
    float4 f1 = *(const float4*)(src + 4);
    union { uint4 q; __bf16 h[8]; } u;
    u.h[0] = (__bf16)f0.x; u.h[1] = (__bf16)f0.y;
    u.h[2] = (__bf16)f0.z; u.h[3] = (__bf16)f0.w;
    u.h[4] = (__bf16)f1.x; u.h[5] = (__bf16)f1.y;
    u.h[6] = (__bf16)f1.z; u.h[7] = (__bf16)f1.w;
    *(uint4*)dst = u.q;
}

DEV void copy8(__bf16* dst, const __bf16* src) {
    *(uint4*)dst = *(const uint4*)src;
}

DEV unsigned lds_addr32(const void* p) {   // LDS aperture: low 32 bits = LDS byte addr
    return (unsigned)(size_t)p;
}

#if HAVE_TDM
// Issue a 2D tile load (bf16, data_size=2B) global->LDS via the Tensor Data
// Mover. D# built per CDNA5 ISA 8.3/8.4: Group0 {count=1, lds_addr,
// global_addr, type=2}; Group1 {data_size=1, pad_enable, pad codes, tensor
// dims, tile dims, dim0 stride}. padi/pada: interval code (3=16dw,4=32dw),
// amount code (3 = 4 dwords = 16B) -> LDS row strides 40/72 elements.
DEV void tdm_load_2d(unsigned lds_off, const void* gptr,
                     unsigned tdim0, unsigned tdim1, unsigned stride,
                     unsigned tile0, unsigned tile1,
                     unsigned padi, unsigned pada) {
    unsigned long long ga = (unsigned long long)gptr;
    v4u g0;
    g0[0] = 1u;                                   // count=1 (valid), user mode
    g0[1] = lds_off;                              // lds_addr (bytes)
    g0[2] = (unsigned)(ga & 0xFFFFFFFFu);         // global_addr[31:0]
    g0[3] = (unsigned)((ga >> 32) & 0x1FFFFFFu) | (2u << 30);  // addr[56:32]|type=2
    v8i g1;
    g1[0] = (int)((1u << 16) | (1u << 20) | (padi << 22) | (pada << 25));
    g1[1] = (int)((tdim0 & 0xFFFFu) << 16);                       // dim0[15:0]
    g1[2] = (int)(((tdim0 >> 16) & 0xFFFFu) | ((tdim1 & 0xFFFFu) << 16));
    g1[3] = (int)(((tdim1 >> 16) & 0xFFFFu) | ((tile0 & 0xFFFFu) << 16));
    g1[4] = (int)(tile1 & 0xFFFFu);                               // tile_dim1; dim2=0
    g1[5] = (int)stride;                                          // dim0_stride lo32
    g1[6] = 0;
    g1[7] = 0;
    v4i z4 = {0, 0, 0, 0};
#if __clang_major__ >= 23
    v8i z8 = {0, 0, 0, 0, 0, 0, 0, 0};
    __builtin_amdgcn_tensor_load_to_lds(g0, g1, z4, z4, z8, 0);
#else
    __builtin_amdgcn_tensor_load_to_lds(g0, g1, z4, z4, 0);
#endif
}
#endif

// ---------------------------------------------------------------------------
// Kernel 0: elementwise f32 -> bf16 (8 elems/thread, all b128 traffic)
// ---------------------------------------------------------------------------
__global__ __launch_bounds__(256)
void cvt_bf16_kernel(const float* __restrict__ src, __bf16* __restrict__ dst, int n8) {
    int i = blockIdx.x * 256 + threadIdx.x;
    if (i < n8) cvt_store8(dst + (size_t)i * 8, src + (size_t)i * 8);
}

// ---------------------------------------------------------------------------
// Kernel 1: per-head projection. Block tile 128x64, 8 waves of 32x32.
// A-tile (x) double-buffered via TDM; B-tile transposed manually.
// ---------------------------------------------------------------------------
__global__ __launch_bounds__(256)
void qkv_proj_kernel(const __bf16* __restrict__ x, const __bf16* __restrict__ W,
                     __bf16* __restrict__ out) {
    __shared__ alignas(16) __bf16 Asm[2][128][40];   // [m][k]
    __shared__ alignas(16) __bf16 BsmT[64][40];      // [n][k]
    const int m0 = blockIdx.x * 128;
    const int h  = blockIdx.y;
    const int tid = threadIdx.x, wave = tid >> 5, lane = tid & 31;
    const int half = lane >> 4, l16 = lane & 15;
    const int wm = wave & 3, wn = wave >> 2;
    constexpr int NK = C / 32;

#if HAVE_TDM
    if (tid == 0)
        tdm_load_2d(lds_addr32(&Asm[0][0][0]), x + (size_t)m0 * C,
                    C, BT, C, 32, 128, 3, 3);
#endif

    v8f c00 = {}, c01 = {}, c10 = {}, c11 = {};

    for (int ks = 0; ks < NK; ++ks) {
        const int k0 = ks * 32;
        const int buf = ks & 1;
        // B tile 32x64 -> BsmT[n][k] (transpose scatter; coalesced global)
        {
            int kk = tid >> 3, nn = (tid & 7) * 8;
            union { uint4 q4; __bf16 hh[8]; } u;
            u.q4 = *(const uint4*)&W[(size_t)h * C * HS + (size_t)(k0 + kk) * HS + nn];
            #pragma unroll
            for (int j = 0; j < 8; ++j) BsmT[nn + j][kk] = u.hh[j];
        }
#if HAVE_TDM
        if (tid == 0) {
            if (ks + 1 < NK) {
                tdm_load_2d(lds_addr32(&Asm[buf ^ 1][0][0]),
                            x + (size_t)m0 * C + (k0 + 32),
                            C, BT, C, 32, 128, 3, 3);
                __builtin_amdgcn_s_wait_tensorcnt(1);
            } else {
                __builtin_amdgcn_s_wait_tensorcnt(0);
            }
        }
#else
        #pragma unroll
        for (int j = 0; j < 2; ++j) {
            int e8 = tid * 2 + j;
            int r = e8 >> 2, cr = (e8 & 3) * 8;
            copy8(&Asm[buf][r][cr], &x[(size_t)(m0 + r) * C + k0 + cr]);
        }
#endif
        __syncthreads();

        v16bf a0 = load_frag(&Asm[buf][wm * 32 + l16][0], half);
        v16bf a1 = load_frag(&Asm[buf][wm * 32 + 16 + l16][0], half);
        v16bf b0 = load_frag(&BsmT[wn * 32 + l16][0], half);
        v16bf b1 = load_frag(&BsmT[wn * 32 + 16 + l16][0], half);
        c00 = wmma_bf16(a0, b0, c00);
        c01 = wmma_bf16(a0, b1, c01);
        c10 = wmma_bf16(a1, b0, c10);
        c11 = wmma_bf16(a1, b1, c11);
        __syncthreads();
    }

    #pragma unroll
    for (int mt = 0; mt < 2; ++mt) {
        const v8f& ca = mt ? c10 : c00;
        const v8f& cb = mt ? c11 : c01;
        #pragma unroll
        for (int r = 0; r < 8; ++r) {
            int m  = m0 + wm * 32 + mt * 16 + r + 8 * half;
            int bb = m >> 11;
            int tt = m & (T - 1);
            size_t base = ((size_t)(bb * H + h) * T + tt) * HS;
            out[base + wn * 32 + l16]      = (__bf16)ca[r];
            out[base + wn * 32 + 16 + l16] = (__bf16)cb[r];
        }
    }
}

// ---------------------------------------------------------------------------
// Kernel 2: causal flash attention; K-tile double-buffered via TDM.
// ---------------------------------------------------------------------------
__global__ __launch_bounds__(256)
void flash_attn_kernel(const __bf16* __restrict__ q, const __bf16* __restrict__ k,
                       const __bf16* __restrict__ v, __bf16* __restrict__ attn) {
    __shared__ alignas(16) __bf16 Ksm[2][32][72];   // [key][feat]
    __shared__ alignas(16) __bf16 VsmT[64][40];     // [d][kv]
    __shared__ alignas(16) __bf16 Psm[8][16][40];   // per-wave [m][kv]

    const int bx = blockIdx.x, h = blockIdx.y, b = blockIdx.z;
    const int tid = threadIdx.x, wave = tid >> 5, lane = tid & 31;
    const int half = lane >> 4, l16 = lane & 15;

    const size_t headbase = (size_t)(b * H + h) * T * HS;
    const __bf16* khead = k + headbase;
    const int qbase = bx * 128 + wave * 16;
    const int qhi   = qbase + 15;

    const __bf16* qr = q + headbase + (size_t)(qbase + l16) * HS;
    v16bf aq0 = load_frag(qr, half);
    v16bf aq1 = load_frag(qr + 32, half);

    v8f o0 = {}, o1 = {}, o2 = {}, o3 = {};
    float mr[8], lr[8];
    #pragma unroll
    for (int r = 0; r < 8; ++r) { mr[r] = -1e30f; lr[r] = 0.f; }

    const int nchunk = (bx * 128 + 128) / 32;
#if HAVE_TDM
    if (tid == 0)
        tdm_load_2d(lds_addr32(&Ksm[0][0][0]), khead, HS, T, HS, 64, 32, 4, 3);
#endif

    for (int ks = 0; ks < nchunk; ++ks) {
        const int kb = ks * 32;
        const int buf = ks & 1;
        // V chunk: coalesced global, transpose-scatter into VsmT[d][kv]
        {
            int kv = tid >> 3, d0 = (tid & 7) * 8;
            union { uint4 q4; __bf16 hh[8]; } u;
            u.q4 = *(const uint4*)&v[headbase + (size_t)(kb + kv) * HS + d0];
            #pragma unroll
            for (int j = 0; j < 8; ++j) VsmT[d0 + j][kv] = u.hh[j];
        }
#if HAVE_TDM
        if (tid == 0) {
            if (ks + 1 < nchunk) {
                tdm_load_2d(lds_addr32(&Ksm[buf ^ 1][0][0]),
                            khead + (size_t)(kb + 32) * HS, HS, T, HS, 64, 32, 4, 3);
                __builtin_amdgcn_s_wait_tensorcnt(1);
            } else {
                __builtin_amdgcn_s_wait_tensorcnt(0);
            }
        }
#else
        {
            int row = tid >> 3, cr = (tid & 7) * 8;
            copy8(&Ksm[buf][row][cr], &khead[(size_t)(kb + row) * HS + cr]);
        }
#endif
        __syncthreads();

        if (kb <= qhi) {                 // wave-uniform causal skip
            v8f s0 = {}, s1 = {};
            s0 = wmma_bf16(aq0, load_frag(&Ksm[buf][l16][0], half), s0);
            s0 = wmma_bf16(aq1, load_frag(&Ksm[buf][l16][32], half), s0);
            s1 = wmma_bf16(aq0, load_frag(&Ksm[buf][16 + l16][0], half), s1);
            s1 = wmma_bf16(aq1, load_frag(&Ksm[buf][16 + l16][32], half), s1);

            const int col0 = kb + l16, col1 = kb + 16 + l16;
            float alpha[8], p0[8], p1[8];
            #pragma unroll
            for (int r = 0; r < 8; ++r) {
                int qrow = qbase + r + 8 * half;
                float x0 = (col0 <= qrow) ? s0[r] * 0.125f : -1e30f;
                float x1 = (col1 <= qrow) ? s1[r] * 0.125f : -1e30f;
                float mv = fmaxf(x0, x1);
                #pragma unroll
                for (int d = 1; d < 16; d <<= 1) mv = fmaxf(mv, __shfl_xor(mv, d, 32));
                float mnew = fmaxf(mr[r], mv);
                alpha[r] = __expf(mr[r] - mnew);
                mr[r] = mnew;
                p0[r] = __expf(x0 - mnew);
                p1[r] = __expf(x1 - mnew);
                float rs = p0[r] + p1[r];
                #pragma unroll
                for (int d = 1; d < 16; d <<= 1) rs += __shfl_xor(rs, d, 32);
                lr[r] = lr[r] * alpha[r] + rs;
            }
            #pragma unroll
            for (int r = 0; r < 8; ++r) {
                o0[r] *= alpha[r]; o1[r] *= alpha[r];
                o2[r] *= alpha[r]; o3[r] *= alpha[r];
                Psm[wave][r + 8 * half][l16]      = (__bf16)p0[r];
                Psm[wave][r + 8 * half][16 + l16] = (__bf16)p1[r];
            }
            asm volatile("s_wait_dscnt 0" ::: "memory");   // intra-wave LDS RAW

            v16bf ap = load_frag(&Psm[wave][l16][0], half);
            o0 = wmma_bf16(ap, load_frag(&VsmT[l16][0], half), o0);
            o1 = wmma_bf16(ap, load_frag(&VsmT[16 + l16][0], half), o1);
            o2 = wmma_bf16(ap, load_frag(&VsmT[32 + l16][0], half), o2);
            o3 = wmma_bf16(ap, load_frag(&VsmT[48 + l16][0], half), o3);
        }
        __syncthreads();
    }

    #pragma unroll
    for (int r = 0; r < 8; ++r) {
        int qrow = qbase + r + 8 * half;
        float inv = 1.0f / lr[r];
        size_t rowbase = ((size_t)b * T + qrow) * C + h * HS;
        attn[rowbase + l16]      = (__bf16)(o0[r] * inv);
        attn[rowbase + 16 + l16] = (__bf16)(o1[r] * inv);
        attn[rowbase + 32 + l16] = (__bf16)(o2[r] * inv);
        attn[rowbase + 48 + l16] = (__bf16)(o3[r] * inv);
    }
}

// ---------------------------------------------------------------------------
// Kernel 3: out = attn @ Wp^T + bp. Both tiles double-buffered via TDM
// (Wp^T already has K contiguous in memory).
// ---------------------------------------------------------------------------
__global__ __launch_bounds__(256)
void out_proj_kernel(const __bf16* __restrict__ A, const __bf16* __restrict__ Wp,
                     const float* __restrict__ bp, float* __restrict__ out) {
    __shared__ alignas(16) __bf16 Asm[2][128][40];
    __shared__ alignas(16) __bf16 Bsm[2][64][40];
    const int m0 = blockIdx.x * 128;
    const int n0 = blockIdx.y * 64;
    const int tid = threadIdx.x, wave = tid >> 5, lane = tid & 31;
    const int half = lane >> 4, l16 = lane & 15;
    const int wm = wave & 3, wn = wave >> 2;
    constexpr int NK = C / 32;

#if HAVE_TDM
    if (tid == 0) {
        tdm_load_2d(lds_addr32(&Asm[0][0][0]), A + (size_t)m0 * C,
                    C, BT, C, 32, 128, 3, 3);
        tdm_load_2d(lds_addr32(&Bsm[0][0][0]), Wp + (size_t)n0 * C,
                    C, C, C, 32, 64, 3, 3);
    }
#endif

    v8f c00 = {}, c01 = {}, c10 = {}, c11 = {};

    for (int ks = 0; ks < NK; ++ks) {
        const int k0 = ks * 32;
        const int buf = ks & 1;
#if HAVE_TDM
        if (tid == 0) {
            if (ks + 1 < NK) {
                tdm_load_2d(lds_addr32(&Asm[buf ^ 1][0][0]),
                            A + (size_t)m0 * C + (k0 + 32), C, BT, C, 32, 128, 3, 3);
                tdm_load_2d(lds_addr32(&Bsm[buf ^ 1][0][0]),
                            Wp + (size_t)n0 * C + (k0 + 32), C, C, C, 32, 64, 3, 3);
                __builtin_amdgcn_s_wait_tensorcnt(2);
            } else {
                __builtin_amdgcn_s_wait_tensorcnt(0);
            }
        }
#else
        #pragma unroll
        for (int j = 0; j < 2; ++j) {
            int e8 = tid * 2 + j;
            int r = e8 >> 2, cr = (e8 & 3) * 8;
            copy8(&Asm[buf][r][cr], &A[(size_t)(m0 + r) * C + k0 + cr]);
        }
        {
            int nn = tid >> 2, kk = (tid & 3) * 8;
            copy8(&Bsm[buf][nn][kk], &Wp[(size_t)(n0 + nn) * C + k0 + kk]);
        }
#endif
        __syncthreads();

        v16bf a0 = load_frag(&Asm[buf][wm * 32 + l16][0], half);
        v16bf a1 = load_frag(&Asm[buf][wm * 32 + 16 + l16][0], half);
        v16bf b0 = load_frag(&Bsm[buf][wn * 32 + l16][0], half);
        v16bf b1 = load_frag(&Bsm[buf][wn * 32 + 16 + l16][0], half);
        c00 = wmma_bf16(a0, b0, c00);
        c01 = wmma_bf16(a0, b1, c01);
        c10 = wmma_bf16(a1, b0, c10);
        c11 = wmma_bf16(a1, b1, c11);
        __syncthreads();
    }

    #pragma unroll
    for (int mt = 0; mt < 2; ++mt) {
        const v8f& ca = mt ? c10 : c00;
        const v8f& cb = mt ? c11 : c01;
        #pragma unroll
        for (int r = 0; r < 8; ++r) {
            int m  = m0 + wm * 32 + mt * 16 + r + 8 * half;
            int na = n0 + wn * 32 + l16;
            int nb = na + 16;
            out[(size_t)m * C + na] = ca[r] + bp[na];
            out[(size_t)m * C + nb] = cb[r] + bp[nb];
        }
    }
}

// ---------------------------------------------------------------------------
extern "C" void kernel_launch(void* const* d_in, const int* in_sizes, int n_in,
                              void* d_out, int out_size, void* d_ws, size_t ws_size,
                              hipStream_t stream) {
    const float* x  = (const float*)d_in[0];
    const float* Wq = (const float*)d_in[1];
    const float* Wk = (const float*)d_in[2];
    const float* Wv = (const float*)d_in[3];
    const float* Wp = (const float*)d_in[4];
    const float* bp = (const float*)d_in[5];
    float* out = (float*)d_out;

    const size_t NX   = (size_t)BT * C;        // 6291456
    const size_t NW   = (size_t)H * C * HS;    // 589824 (== C*C)
    const size_t NQKV = (size_t)B * H * T * HS;

    __bf16* xbf  = (__bf16*)d_ws;
    __bf16* wqbf = xbf + NX;
    __bf16* wkbf = wqbf + NW;
    __bf16* wvbf = wkbf + NW;
    __bf16* wpbf = wvbf + NW;
    __bf16* qb   = wpbf + NW;
    __bf16* kb   = qb + NQKV;
    __bf16* vb   = kb + NQKV;
    __bf16* ab   = vb + NQKV;

    dim3 blk(256);
    // pre-convert everything to bf16 once
    hipLaunchKernelGGL(cvt_bf16_kernel, dim3((NX / 8 + 255) / 256), blk, 0, stream, x,  xbf,  (int)(NX / 8));
    hipLaunchKernelGGL(cvt_bf16_kernel, dim3((NW / 8 + 255) / 256), blk, 0, stream, Wq, wqbf, (int)(NW / 8));
    hipLaunchKernelGGL(cvt_bf16_kernel, dim3((NW / 8 + 255) / 256), blk, 0, stream, Wk, wkbf, (int)(NW / 8));
    hipLaunchKernelGGL(cvt_bf16_kernel, dim3((NW / 8 + 255) / 256), blk, 0, stream, Wv, wvbf, (int)(NW / 8));
    hipLaunchKernelGGL(cvt_bf16_kernel, dim3((NW / 8 + 255) / 256), blk, 0, stream, Wp, wpbf, (int)(NW / 8));

    dim3 gproj(BT / 128, H);
    hipLaunchKernelGGL(qkv_proj_kernel, gproj, blk, 0, stream, xbf, wqbf, qb);
    hipLaunchKernelGGL(qkv_proj_kernel, gproj, blk, 0, stream, xbf, wkbf, kb);
    hipLaunchKernelGGL(qkv_proj_kernel, gproj, blk, 0, stream, xbf, wvbf, vb);

    dim3 gattn(T / 128, H, B);
    hipLaunchKernelGGL(flash_attn_kernel, gattn, blk, 0, stream, qb, kb, vb, ab);

    dim3 gout(BT / 128, C / 64);
    hipLaunchKernelGGL(out_proj_kernel, gout, blk, 0, stream, ab, wpbf, bp, out);
}
